// AdaGATConv_76166950028494
// MI455X (gfx1250) — compile-verified, matrix operand order checked by reading, not run on hardware
//
#include <hip/hip_runtime.h>
#include <hip/hip_bf16.h>

#define N_NODES 20000
#define N_EDGES 320000
#define H 4
#define F 64
#define IN_DIM 256

typedef __attribute__((ext_vector_type(16))) __bf16        v16bf;
typedef __attribute__((ext_vector_type(8)))  float         v8f;
typedef __attribute__((ext_vector_type(8)))  unsigned int  vu8;   // 32B = one bf16 fragment

__device__ __forceinline__ unsigned short bf16bits(float f) {
    unsigned u = __builtin_bit_cast(unsigned, f);
    return (unsigned short)((u + 0x7FFFu + ((u >> 16) & 1u)) >> 16);   // RNE
}
__device__ __forceinline__ __bf16 f2bf(float f) {
    return __builtin_bit_cast(__bf16, bf16bits(f));
}
__device__ __forceinline__ float leaky(float x) { return x >= 0.f ? x : 0.2f * x; }
// monotonic float<->uint encoding for atomicMax on floats
__device__ __forceinline__ unsigned fenc(float f) {
    unsigned u = __builtin_bit_cast(unsigned, f);
    return (u & 0x80000000u) ? ~u : (u | 0x80000000u);
}
__device__ __forceinline__ float fdec(unsigned u) {
    unsigned b = (u & 0x80000000u) ? (u & 0x7FFFFFFFu) : ~u;
    return __builtin_bit_cast(float, b);
}

// ---------------------------------------------------------------------------
// Repack f32 B[256, Ncols] into bf16 WMMA B-fragment layout:
//   vu8 Bp[(tileN*8 + kstep)*32 + lane] = 16 halves, half i at
//   k = kstep*32 + (i/8)*16 + (lane>=16 ? 8:0) + i%8, col = tileN*16 + lane%16
// One thread per (tileN, kstep, lane).
// ---------------------------------------------------------------------------
__global__ void repackB_kernel(const float* __restrict__ B, int ldb, int tilesN,
                               vu8* __restrict__ Bp) {
    const int idx = blockIdx.x * blockDim.x + threadIdx.x;
    if (idx >= tilesN * 8 * 32) return;
    const int lane = idx & 31, ks = (idx >> 5) & 7, tn = idx >> 8;
    const int col = tn * 16 + (lane & 15);
    const int hi  = lane >> 4;
    vu8 out;
#pragma unroll
    for (int p = 0; p < 8; ++p) {
        const int i0 = 2 * p, i1 = 2 * p + 1;
        const int k0 = ks * 32 + ((i0 >> 3) << 4) + (hi << 3) + (i0 & 7);
        const int k1 = ks * 32 + ((i1 >> 3) << 4) + (hi << 3) + (i1 & 7);
        out[p] = (unsigned)bf16bits(B[k0 * ldb + col]) |
                 ((unsigned)bf16bits(B[k1 * ldb + col]) << 16);
    }
    Bp[idx] = out;
}

// ---------------------------------------------------------------------------
// bf16 WMMA GEMM: C[M, Ncols] = A[M,256] * B, fp32 accum.
// A: 4 K-segment pointers (segment s covers k in [64s, 64s+64)).
// B: pre-packed fragment layout (repackB_kernel).
// Each wave computes TN adjacent 16x16 N-tiles per A fragment (A reuse).
// K fixed = 256 (8 x v_wmma_f32_16x16x32_bf16 per N-tile).
// ---------------------------------------------------------------------------
template <int TN>
__global__ void gemm_wmma(const float* __restrict__ A0, const float* __restrict__ A1,
                          const float* __restrict__ A2, const float* __restrict__ A3,
                          int lda, const vu8* __restrict__ Bp,
                          float* __restrict__ C, int ldc, int tilesM, int nGroups) {
    const int wave = threadIdx.x >> 5;
    const int lane = threadIdx.x & 31;
    const int t = blockIdx.x * 8 + wave;
    if (t >= tilesM * nGroups) return;                 // wave-uniform exit
    const int tm = t / nGroups, tng = t % nGroups;
    const float* Aseg[4] = {A0, A1, A2, A3};

    const int lo  = lane & 15;
    const int hi  = lane >> 4;
    const int row = tm * 16 + lo;

    v8f acc[TN];
#pragma unroll
    for (int j = 0; j < TN; ++j) acc[j] = (v8f){0.f, 0.f, 0.f, 0.f, 0.f, 0.f, 0.f, 0.f};

    for (int kk = 0; kk < 256; kk += 32) {
        // ---- A fragment: two 8-float runs, 16B-aligned float4 loads ----
        const float* Ap = Aseg[kk >> 6] + (size_t)row * lda + (kk & 63) + (hi << 3);
        const float4 x0 = *(const float4*)(Ap);
        const float4 x1 = *(const float4*)(Ap + 4);
        const float4 x2 = *(const float4*)(Ap + 16);
        const float4 x3 = *(const float4*)(Ap + 20);
        v16bf a;
        a[0] = f2bf(x0.x);  a[1] = f2bf(x0.y);  a[2]  = f2bf(x0.z);  a[3]  = f2bf(x0.w);
        a[4] = f2bf(x1.x);  a[5] = f2bf(x1.y);  a[6]  = f2bf(x1.z);  a[7]  = f2bf(x1.w);
        a[8] = f2bf(x2.x);  a[9] = f2bf(x2.y);  a[10] = f2bf(x2.z);  a[11] = f2bf(x2.w);
        a[12] = f2bf(x3.x); a[13] = f2bf(x3.y); a[14] = f2bf(x3.z);  a[15] = f2bf(x3.w);

        const int ks = kk >> 5;
#pragma unroll
        for (int j = 0; j < TN; ++j) {
            const int tn = tng * TN + j;
            const v16bf b = __builtin_bit_cast(v16bf, Bp[(tn * 8 + ks) * 32 + lane]);
            acc[j] = __builtin_amdgcn_wmma_f32_16x16x32_bf16(false, a, false, b,
                                                             (short)0, acc[j], false, false);
        }
    }
    // C layout: VGPR r -> m = tm*16 + r + 8*hi, n = tn*16 + lo
    const int m0 = tm * 16 + (hi << 3);
#pragma unroll
    for (int j = 0; j < TN; ++j) {
        const int n = (tng * TN + j) * 16 + lo;
#pragma unroll
        for (int r = 0; r < 8; ++r) C[(size_t)(m0 + r) * ldc + n] = acc[j][r];
    }
}

// ---------------------------------------------------------------------------
// Fold small weights on device (1 block, 256 threads; thread = (k,h))
//   w2l   = fold(W_fc2, attn_l)              [64,4]
//   wEl   = (I + WA) @ w2l                   [64,4]  (node-side logit weight)
//   Bedge [256,16]: rows 0..63  cols 0..3  = W_fc0 @ (WB @ w2l)
//                   rows 64..   cols 4..7  = mean-fold(W_ada_c)
//                   rows 128..  cols 8..11 = mean-fold(W_ada_t)
//                   rows 192..  cols 12..15= mean-fold(W_ada_d), zeros elsewhere
// ---------------------------------------------------------------------------
__global__ void fold_kernel(const float* __restrict__ W_fc2, const float* __restrict__ attn_l,
                            const float* __restrict__ W_fc_src, const float* __restrict__ W_fc0,
                            const float* __restrict__ Wac, const float* __restrict__ Wat,
                            const float* __restrict__ Wad,
                            float* __restrict__ wEl, float* __restrict__ Bedge) {
    __shared__ float s_w2l[64 * 4];
    __shared__ float s_wB2l[64 * 4];
    const int tid = threadIdx.x, k = tid >> 2, h = tid & 3;
    for (int i = tid; i < 256 * 16; i += 256) Bedge[i] = 0.f;
    float acc = 0.f;
    for (int f = 0; f < 64; ++f) acc += W_fc2[k * 256 + h * 64 + f] * attn_l[h * 64 + f];
    s_w2l[k * 4 + h] = acc;
    __syncthreads();
    float aA = 0.f, aB = 0.f;
    for (int j = 0; j < 64; ++j) {
        const float w = s_w2l[j * 4 + h];
        aA += W_fc_src[k * 64 + j] * w;             // WA row k
        aB += W_fc_src[(64 + k) * 64 + j] * w;      // WB row k
    }
    wEl[k * 4 + h]    = s_w2l[k * 4 + h] + aA;
    s_wB2l[k * 4 + h] = aB;
    __syncthreads();
    float aE = 0.f;
    for (int j = 0; j < 64; ++j) aE += W_fc0[k * 64 + j] * s_wB2l[j * 4 + h];
    Bedge[k * 16 + h] = aE;
    float mc = 0.f, mt = 0.f, md = 0.f;
    for (int f = 0; f < 64; ++f) {
        mc += Wac[k * 256 + h * 64 + f];
        mt += Wat[k * 256 + h * 64 + f];
        md += Wad[k * 256 + h * 64 + f];
    }
    Bedge[(64  + k) * 16 + 4  + h] = mc * (1.f / 64.f);
    Bedge[(128 + k) * 16 + 8  + h] = mt * (1.f / 64.f);
    Bedge[(192 + k) * 16 + 12 + h] = md * (1.f / 64.f);
}

__global__ void init_kernel(float* __restrict__ out, const float* __restrict__ bias,
                            unsigned* __restrict__ menc, float* __restrict__ den) {
    const int idx = blockIdx.x * blockDim.x + threadIdx.x;
    if (idx < N_NODES * H * F) out[idx] = bias[idx & 255];
    if (idx < N_NODES * H) { menc[idx] = 0x007FFFFFu; /* enc(-inf) */ den[idx] = 0.f; }
}

__global__ void ft_kernel(const float* __restrict__ ftp, float* __restrict__ ft) {
    const int idx = blockIdx.x * blockDim.x + threadIdx.x;
    if (idx < N_NODES * F) ft[idx] = leaky(ftp[idx]);
}

// per (node, head): er = sum_f leaky(fd)*attn_r ;  elA = ft . wEl[:,h]
__global__ void node_kernel(const float* __restrict__ fd, const float* __restrict__ ft,
                            const float* __restrict__ attn_r, const float* __restrict__ wEl,
                            float* __restrict__ er, float* __restrict__ elA) {
    const int idx = blockIdx.x * blockDim.x + threadIdx.x;
    if (idx >= N_NODES * H) return;
    const int n = idx >> 2, h = idx & 3;
    float e = 0.f, a = 0.f;
    for (int f = 0; f < 64; ++f) e += leaky(fd[n * 256 + h * 64 + f]) * attn_r[h * 64 + f];
    for (int k = 0; k < 64; ++k) a += ft[n * 64 + k] * wEl[k * 4 + h];
    er[idx] = e;
    elA[idx] = a;
}

// per (edge, head): logit + segment atomic-max
__global__ void edge_logit_kernel(const float* __restrict__ eout, const float* __restrict__ elA,
                                  const float* __restrict__ er, const int* __restrict__ src,
                                  const int* __restrict__ dst, const float* __restrict__ pac,
                                  const float* __restrict__ pat, const float* __restrict__ pad_,
                                  float* __restrict__ logit, unsigned* __restrict__ menc) {
    const int idx = blockIdx.x * blockDim.x + threadIdx.x;
    if (idx >= N_EDGES * H) return;
    const int e = idx >> 2, h = idx & 3;
    const int d = dst[e], s = src[e];
    const float el = eout[e * 16 + h] + elA[s * 4 + h] + er[d * 4 + h];
    const float dec = pac[0] * eout[e * 16 + 4 + h] + pat[0] * eout[e * 16 + 8 + h] +
                      pad_[0] * eout[e * 16 + 12 + h];
    const float lg = leaky(el * expf(-dec));
    logit[idx] = lg;
    atomicMax(&menc[d * 4 + h], fenc(lg));
}

// per (edge, head): ex = exp(logit - m[dst]); den[dst] += ex   (in-place on logit)
__global__ void edge_exp_kernel(const int* __restrict__ dst, const unsigned* __restrict__ menc,
                                float* __restrict__ logit, float* __restrict__ den) {
    const int idx = blockIdx.x * blockDim.x + threadIdx.x;
    if (idx >= N_EDGES * H) return;
    const int e = idx >> 2, h = idx & 3;
    const float ex = expf(logit[idx] - fdec(menc[dst[e] * 4 + h]));
    logit[idx] = ex;
    atomicAdd(&den[dst[e] * 4 + h], ex);
}

// per (edge, head): a = ex/den[dst]; a<1e-5 -> 0   (in-place on logit)
__global__ void edge_norm_kernel(const int* __restrict__ dst, const float* __restrict__ den,
                                 float* __restrict__ logit) {
    const int idx = blockIdx.x * blockDim.x + threadIdx.x;
    if (idx >= N_EDGES * H) return;
    const int e = idx >> 2, h = idx & 3;
    float a = logit[idx] / den[dst[e] * 4 + h];
    if (a < 1e-5f) a = 0.f;
    logit[idx] = a;
}

// per (edge, f): rst[dst, h, f] += ft[src, f] * a[e, h]
__global__ void scatter_kernel(const int* __restrict__ src, const int* __restrict__ dst,
                               const float* __restrict__ ft, const float* __restrict__ a,
                               float* __restrict__ out) {
    const int idx = blockIdx.x * blockDim.x + threadIdx.x;
    if (idx >= N_EDGES * F) return;
    const int e = idx >> 6, f = idx & 63;
    const int d = dst[e];
    const float fv = ft[src[e] * 64 + f];
#pragma unroll
    for (int h = 0; h < 4; ++h) {
        const float ah = a[e * 4 + h];
        if (ah != 0.f) atomicAdd(&out[d * 256 + h * 64 + f], fv * ah);
    }
}

extern "C" void kernel_launch(void* const* d_in, const int* in_sizes, int n_in,
                              void* d_out, int out_size, void* d_ws, size_t ws_size,
                              hipStream_t stream) {
    const float* feat     = (const float*)d_in[0];
    const float* edg      = (const float*)d_in[1];
    const float* ada_c    = (const float*)d_in[2];
    const float* ada_t    = (const float*)d_in[3];
    const float* ada_d    = (const float*)d_in[4];
    const int*   src      = (const int*)d_in[5];
    const int*   dst      = (const int*)d_in[6];
    const float* W_fc     = (const float*)d_in[7];
    const float* W_fc0    = (const float*)d_in[8];
    const float* W_fc1    = (const float*)d_in[9];
    const float* W_fc2    = (const float*)d_in[10];
    const float* W_fc_src = (const float*)d_in[11];
    const float* W_ada_c  = (const float*)d_in[12];
    const float* W_ada_t  = (const float*)d_in[13];
    const float* W_ada_d  = (const float*)d_in[14];
    const float* a_c      = (const float*)d_in[15];
    const float* a_t      = (const float*)d_in[16];
    const float* a_d      = (const float*)d_in[17];
    const float* attn_l   = (const float*)d_in[18];
    const float* attn_r   = (const float*)d_in[19];
    const float* bias     = (const float*)d_in[20];
    float* out = (float*)d_out;

    char* w = (char*)d_ws;
    float* fd    = (float*)w; w += (size_t)N_NODES * 256 * 4;  // feat@W_fc ; aliased as eout [E,16]
    float* ftp   = (float*)w; w += (size_t)N_NODES * 64 * 4;   // feat@W_fc1 (pre-leaky)
    float* ft    = (float*)w; w += (size_t)N_NODES * 64 * 4;   // leaky(ftp)
    float* er    = (float*)w; w += (size_t)N_NODES * 4 * 4;
    float* elA   = (float*)w; w += (size_t)N_NODES * 4 * 4;
    float* den   = (float*)w; w += (size_t)N_NODES * 4 * 4;
    unsigned* menc = (unsigned*)w; w += (size_t)N_NODES * 4 * 4;
    float* wEl   = (float*)w; w += 256 * 4;
    float* Bedge = (float*)w; w += 256 * 16 * 4;
    float* logit = (float*)w; w += (size_t)N_EDGES * 4 * 4;
    vu8* Bp_fc   = (vu8*)w;   w += (size_t)16 * 8 * 32 * 32;   // packed W_fc   (16 tiles)
    vu8* Bp_fc1  = (vu8*)w;   w += (size_t)4  * 8 * 32 * 32;   // packed W_fc1  (4 tiles)
    vu8* Bp_edge = (vu8*)w;   w += (size_t)1  * 8 * 32 * 32;   // packed Bedge  (1 tile)
    float* eout  = fd;                                         // [E,16] == [N,256] in size

    // 1. fold small weights, pack B matrices into WMMA fragment layout
    fold_kernel<<<1, 256, 0, stream>>>(W_fc2, attn_l, W_fc_src, W_fc0,
                                       W_ada_c, W_ada_t, W_ada_d, wEl, Bedge);
    repackB_kernel<<<(16 * 256 + 255) / 256, 256, 0, stream>>>(W_fc, 256, 16, Bp_fc);
    repackB_kernel<<<(4 * 256 + 255) / 256, 256, 0, stream>>>(W_fc1, 64, 4, Bp_fc1);
    repackB_kernel<<<1, 256, 0, stream>>>(Bedge, 16, 1, Bp_edge);
    // 2. init output (bias) + softmax state
    init_kernel<<<(N_NODES * 256 + 255) / 256, 256, 0, stream>>>(out, bias, menc, den);
    // 3. fd = feat @ W_fc  (20000x256x256), 4 N-tiles per wave
    gemm_wmma<4><<<(1250 * 4 + 7) / 8, 256, 0, stream>>>(
        feat, feat + 64, feat + 128, feat + 192, 256, Bp_fc, fd, 256, 1250, 4);
    // 4. ftp = feat @ W_fc1  (20000x256x64), 4 N-tiles per wave
    gemm_wmma<4><<<(1250 * 1 + 7) / 8, 256, 0, stream>>>(
        feat, feat + 64, feat + 128, feat + 192, 256, Bp_fc1, ftp, 64, 1250, 1);
    // 5. ft = leaky(ftp)
    ft_kernel<<<(N_NODES * 64 + 255) / 256, 256, 0, stream>>>(ftp, ft);
    // 6. er, elA per (node, head)
    node_kernel<<<(N_NODES * 4 + 255) / 256, 256, 0, stream>>>(fd, ft, attn_r, wEl, er, elA);
    // 7. eout = [edg|ada_c|ada_t|ada_d] @ Bedge  (320000x256x16)
    gemm_wmma<1><<<(20000 + 7) / 8, 256, 0, stream>>>(
        edg, ada_c, ada_t, ada_d, 64, Bp_edge, eout, 16, 20000, 1);
    // 8. logits + segment max
    edge_logit_kernel<<<(N_EDGES * 4 + 255) / 256, 256, 0, stream>>>(
        eout, elA, er, src, dst, a_c, a_t, a_d, logit, menc);
    // 9. exp + segment sum
    edge_exp_kernel<<<(N_EDGES * 4 + 255) / 256, 256, 0, stream>>>(dst, menc, logit, den);
    // 10. normalize + threshold
    edge_norm_kernel<<<(N_EDGES * 4 + 255) / 256, 256, 0, stream>>>(dst, den, logit);
    // 11. message scatter
    scatter_kernel<<<(N_EDGES * 64 + 255) / 256, 256, 0, stream>>>(src, dst, ft, logit, out);
}
